// ClusterModule_48180943127372
// MI455X (gfx1250) — matrix operand order repaired and checked by reading, not compiled, not run on hardware
//
#include <hip/hip_runtime.h>
#include <hip/hip_bf16.h>

#define BATCH   32
#define KPTS    4096
#define CFEAT   256
#define NPOINT  256
#define NSAMPLE 16
#define RADIUS  0.3f
#define KPAD1   288   // 259 padded to 9 chunks of 32 (xyz cols moved to 256..258)

typedef __attribute__((ext_vector_type(16))) _Float16 v16h;
typedef __attribute__((ext_vector_type(8)))  _Float16 v8h;
typedef __attribute__((ext_vector_type(8)))  float    v8f;

// ---------------------------------------------------------------------------
// Kernel 1: farthest point sampling. One block per batch, 256 threads,
// each thread owns 16 points in registers (coords + running min-dist).
// ---------------------------------------------------------------------------
__global__ __launch_bounds__(256) void fps_kernel(const float* __restrict__ xyz,
                                                  float* __restrict__ outXYZ,
                                                  float* __restrict__ outInds)
{
    const int b = blockIdx.x;
    const int t = threadIdx.x;
    const float* pts = xyz + (size_t)b * KPTS * 3;

    float px[16], py[16], pz[16], dl[16];
#pragma unroll
    for (int i = 0; i < 16; ++i) {
        int k = t + 256 * i;
        px[i] = pts[3 * k + 0];
        py[i] = pts[3 * k + 1];
        pz[i] = pts[3 * k + 2];
        dl[i] = 1e10f;
    }

    __shared__ float s_val[8];
    __shared__ int   s_idx[8];
    __shared__ float s_f[3];
    __shared__ int   s_far;

    int far = 0;
    for (int it = 0; it < NPOINT; ++it) {
        // owner of `far` publishes its coords and the outputs for this step
        if (t == (far & 255)) {
            int slot = far >> 8;
            s_f[0] = px[slot]; s_f[1] = py[slot]; s_f[2] = pz[slot];
            outXYZ[((size_t)b * NPOINT + it) * 3 + 0] = px[slot];
            outXYZ[((size_t)b * NPOINT + it) * 3 + 1] = py[slot];
            outXYZ[((size_t)b * NPOINT + it) * 3 + 2] = pz[slot];
            outInds[(size_t)b * NPOINT + it] = (float)far;
        }
        __syncthreads();
        float fx = s_f[0], fy = s_f[1], fz = s_f[2];

        float bestv = -1.0f; int besti = 0x7fffffff;
#pragma unroll
        for (int i = 0; i < 16; ++i) {
            float dx = px[i] - fx, dy = py[i] - fy, dz = pz[i] - fz;
            float d = dx * dx + dy * dy + dz * dz;
            float nd = fminf(dl[i], d);
            dl[i] = nd;
            int k = t + 256 * i;
            if (nd > bestv || (nd == bestv && k < besti)) { bestv = nd; besti = k; }
        }
        // wave32 butterfly argmax (tie-break: lowest index, matching argmax)
#pragma unroll
        for (int off = 16; off; off >>= 1) {
            float ov = __shfl_xor(bestv, off, 32);
            int   oi = __shfl_xor(besti, off, 32);
            if (ov > bestv || (ov == bestv && oi < besti)) { bestv = ov; besti = oi; }
        }
        if ((t & 31) == 0) { s_val[t >> 5] = bestv; s_idx[t >> 5] = besti; }
        __syncthreads();
        if (t == 0) {
            float bv = s_val[0]; int bi = s_idx[0];
            for (int w = 1; w < 8; ++w)
                if (s_val[w] > bv || (s_val[w] == bv && s_idx[w] < bi)) { bv = s_val[w]; bi = s_idx[w]; }
            s_far = bi;
        }
        __syncthreads();
        far = s_far;
    }
}

// ---------------------------------------------------------------------------
// Kernel 2: ball query. One block per batch (xyz staged in 48KB LDS),
// one thread per query point; serial scan keeps first-in-index-order
// semantics of the reference, early exit at 16 hits, pad with first hit.
// ---------------------------------------------------------------------------
__global__ __launch_bounds__(256) void ballq_kernel(const float* __restrict__ xyz,
                                                    const float* __restrict__ newXYZ,
                                                    int* __restrict__ ballIdx)
{
    const int b = blockIdx.x;
    const int t = threadIdx.x;
    __shared__ float sp[KPTS * 3];           // 48 KB
    const float* pts = xyz + (size_t)b * KPTS * 3;
    for (int i = t; i < KPTS * 3; i += 256) sp[i] = pts[i];
    __syncthreads();

    const int q = b * NPOINT + t;
    const float qx = newXYZ[(size_t)q * 3 + 0];
    const float qy = newXYZ[(size_t)q * 3 + 1];
    const float qz = newXYZ[(size_t)q * 3 + 2];
    int* out = ballIdx + (size_t)q * NSAMPLE;

    int cnt = 0, first = 0;
    const float r2 = RADIUS * RADIUS;
    for (int k = 0; k < KPTS; ++k) {
        float dx = sp[3 * k + 0] - qx;
        float dy = sp[3 * k + 1] - qy;
        float dz = sp[3 * k + 2] - qz;
        if (dx * dx + dy * dy + dz * dz < r2) {
            if (cnt == 0) first = k;
            out[cnt++] = k;
            if (cnt == NSAMPLE) break;
        }
    }
    for (int j = cnt; j < NSAMPLE; ++j) out[j] = first;
}

// ---------------------------------------------------------------------------
// Kernel T: features (B,C,K) f32  ->  featT (B,K,C) f16 so per-sample feature
// gathers become contiguous 512B rows.
// ---------------------------------------------------------------------------
__global__ __launch_bounds__(256) void transpose_kernel(const float* __restrict__ feat,
                                                        _Float16* __restrict__ featT)
{
    __shared__ float tile[32][33];
    const int b  = blockIdx.z;
    const int k0 = blockIdx.x * 32;
    const int c0 = blockIdx.y * 32;
    const int tx = threadIdx.x, ty = threadIdx.y;
    const float* src = feat + ((size_t)b * CFEAT + c0) * KPTS + k0;
#pragma unroll
    for (int i = 0; i < 32; i += 8) tile[ty + i][tx] = src[(size_t)(ty + i) * KPTS + tx];
    __syncthreads();
    _Float16* dst = featT + ((size_t)b * KPTS + k0) * CFEAT + c0;
#pragma unroll
    for (int i = 0; i < 32; i += 8) dst[(size_t)(ty + i) * CFEAT + tx] = (_Float16)tile[tx][ty + i];
}

// ---------------------------------------------------------------------------
// Kernel W: convert weights to f16 WMMA-friendly layout.
// W1h is 128 x 288: cols 0..255 = feature weights (W1[:,3:259]),
// cols 256..258 = xyz weights (W1[:,0:3]), cols 259..287 = 0.
// ---------------------------------------------------------------------------
__global__ __launch_bounds__(256) void wconv_kernel(const float* __restrict__ W1,
                                                    const float* __restrict__ W2,
                                                    const float* __restrict__ W3,
                                                    _Float16* __restrict__ W1h,
                                                    _Float16* __restrict__ W2h,
                                                    _Float16* __restrict__ W3h)
{
    const int t = blockIdx.x * 256 + threadIdx.x;
    const int stride = gridDim.x * 256;
    for (int i = t; i < 128 * KPAD1; i += stride) {
        int o = i / KPAD1, j = i % KPAD1;
        float v = (j < 256) ? W1[o * 259 + 3 + j]
                            : ((j < 259) ? W1[o * 259 + (j - 256)] : 0.0f);
        W1h[i] = (_Float16)v;
    }
    for (int i = t; i < 128 * 128; i += stride) {
        W2h[i] = (_Float16)W2[i];
        W3h[i] = (_Float16)W3[i];
    }
}

// ---------------------------------------------------------------------------
// Kernel 3: gather + 3-layer MLP (WMMA) + max-pool. One wave per query.
// ---------------------------------------------------------------------------
__device__ __forceinline__ v16h load_a(const _Float16* Xl, int pitch, int kbase,
                                       int m, int hi)
{
    // ISA 16-bit A 16x32 layout: lanes 0-15 hold K {0-7,16-23}, lanes 16-31 {8-15,24-31}
    union { v16h v; v8h h[2]; } u;
    u.h[0] = *(const v8h*)(Xl + m * pitch + kbase + hi * 8);
    u.h[1] = *(const v8h*)(Xl + m * pitch + kbase + 16 + hi * 8);
    return u.v;
}

__device__ __forceinline__ v16h load_b(const _Float16* __restrict__ W, int kw,
                                       int col, int kbase, int hi)
{
    // B 32x16: lane = column (mod 16); lanes 0-15 K {0-7,16-23}, lanes 16-31 {8-15,24-31}
    union { v16h v; v8h h[2]; } u;
    u.h[0] = *(const v8h*)(W + (size_t)col * kw + kbase + hi * 8);
    u.h[1] = *(const v8h*)(W + (size_t)col * kw + kbase + 16 + hi * 8);
    return u.v;
}

// One MLP layer: Y(16x128) = relu((X(16x32*NK) @ W^T) * g + b).
// All NK A-fragments resident in registers; per output tile the NK B-fragments
// are loaded as one clause BEFORE the WMMA chain so loads pipeline with math.
template<int NK>
__device__ __forceinline__ void mlp_stage(const _Float16* Xl, int pitch,
                                          const _Float16* __restrict__ Wg, int kw,
                                          const float* __restrict__ g,
                                          const float* __restrict__ bb,
                                          _Float16* Ylds, float* outp, int lane)
{
    const int m = lane & 15, hi = lane >> 4;

    v16h a[NK];
#pragma unroll
    for (int kc = 0; kc < NK; ++kc) a[kc] = load_a(Xl, pitch, kc * 32, m, hi);

    for (int n0 = 0; n0 < 128; n0 += 16) {
        const int col = n0 + m;
        // issue epilogue scalars + all B fragments up front (one load clause)
        const float gg = g[col], bv = bb[col];
        v16h bf[NK];
#pragma unroll
        for (int kc = 0; kc < NK; ++kc) bf[kc] = load_b(Wg, kw, col, kc * 32, hi);

        v8f acc = {};
#pragma unroll
        for (int kc = 0; kc < NK; ++kc)
            acc = __builtin_amdgcn_wmma_f32_16x16x32_f16(
                false, a[kc], false, bf[kc], (short)0, acc, false, false);

        if (Ylds) {
#pragma unroll
            for (int r = 0; r < 8; ++r) {
                float y = acc[r] * gg + bv;
                y = y > 0.f ? y : 0.f;
                Ylds[(hi * 8 + r) * 128 + col] = (_Float16)y;
            }
        } else {
            float mx = 0.f;  // all post-ReLU values >= 0
#pragma unroll
            for (int r = 0; r < 8; ++r) {
                float y = acc[r] * gg + bv;
                y = y > 0.f ? y : 0.f;
                mx = fmaxf(mx, y);
            }
            mx = fmaxf(mx, __shfl_xor(mx, 16, 32));
            if (hi == 0) outp[col] = mx;
        }
    }
}

__global__ __launch_bounds__(64) void mlp_kernel(
    const float* __restrict__ xyz, const float* __restrict__ features,
    const _Float16* __restrict__ featT, int useT,
    const float* __restrict__ newXYZ, const int* __restrict__ ballIdx,
    const _Float16* __restrict__ W1h, const float* __restrict__ g1, const float* __restrict__ b1,
    const _Float16* __restrict__ W2h, const float* __restrict__ g2, const float* __restrict__ b2,
    const _Float16* __restrict__ W3h, const float* __restrict__ g3, const float* __restrict__ b3,
    float* __restrict__ outFeat)
{
    const int wave = threadIdx.x >> 5;
    const int lane = threadIdx.x & 31;
    const int q = blockIdx.x * 2 + wave;
    const int b = q / NPOINT;

    // warm the weight stream (global_prefetch_b8); L2-hot for every block after
    __builtin_prefetch(W1h + (size_t)lane * 1152, 0, 0);
    __builtin_prefetch(W2h + (size_t)lane * 512, 0, 0);
    __builtin_prefetch(W3h + (size_t)lane * 512, 0, 0);

    __shared__ __align__(16) _Float16 sX [2][16 * KPAD1]; // 2 x 9216 B
    __shared__ __align__(16) _Float16 sY1[2][16 * 128];   // 2 x 4096 B
    __shared__ __align__(16) _Float16 sY2[2][16 * 128];   // 2 x 4096 B
    _Float16* X  = sX[wave];
    _Float16* Y1 = sY1[wave];
    _Float16* Y2 = sY2[wave];

    const float qx = newXYZ[(size_t)q * 3 + 0];
    const float qy = newXYZ[(size_t)q * 3 + 1];
    const float qz = newXYZ[(size_t)q * 3 + 2];
    const int* idx = ballIdx + (size_t)q * NSAMPLE;

    // ---- build X (16 x 288 f16): cols 0..255 features, 256..258 norm-xyz, pad 0
    for (int s = 0; s < NSAMPLE; ++s) {
        const int k = idx[s];
        if (useT) {
            const _Float16* row = featT + ((size_t)b * KPTS + k) * CFEAT;
            *(v8h*)(X + s * KPAD1 + lane * 8) = *(const v8h*)(row + lane * 8);
        } else {
            const float* fb = features + (size_t)b * CFEAT * KPTS + k;
#pragma unroll
            for (int i = 0; i < 8; ++i) {
                int c = lane + 32 * i;
                X[s * KPAD1 + c] = (_Float16)fb[(size_t)c * KPTS];
            }
        }
    }
    if (lane < NSAMPLE) {
        const int k = idx[lane];
        const float* pp = xyz + ((size_t)b * KPTS + k) * 3;
        X[lane * KPAD1 + 256] = (_Float16)((pp[0] - qx) * (1.0f / RADIUS));
        X[lane * KPAD1 + 257] = (_Float16)((pp[1] - qy) * (1.0f / RADIUS));
        X[lane * KPAD1 + 258] = (_Float16)((pp[2] - qz) * (1.0f / RADIUS));
        for (int j = 259; j < KPAD1; ++j) X[lane * KPAD1 + j] = (_Float16)0.0f;
    }
    // same-wave LDS ops are in-order; waves use disjoint LDS regions -> no barrier

    mlp_stage<9>(X,  KPAD1, W1h, KPAD1, g1, b1, Y1, nullptr, lane);
    mlp_stage<4>(Y1, 128,   W2h, 128,   g2, b2, Y2, nullptr, lane);
    mlp_stage<4>(Y2, 128,   W3h, 128,   g3, b3, nullptr,
                 outFeat + (size_t)q * 128, lane);
}

// ---------------------------------------------------------------------------
extern "C" void kernel_launch(void* const* d_in, const int* in_sizes, int n_in,
                              void* d_out, int out_size, void* d_ws, size_t ws_size,
                              hipStream_t stream)
{
    (void)in_sizes; (void)n_in; (void)out_size;
    const float* xyz      = (const float*)d_in[0];
    const float* features = (const float*)d_in[1];
    const float* W1 = (const float*)d_in[2];
    const float* g1 = (const float*)d_in[3];
    const float* b1 = (const float*)d_in[4];
    const float* W2 = (const float*)d_in[5];
    const float* g2 = (const float*)d_in[6];
    const float* b2 = (const float*)d_in[7];
    const float* W3 = (const float*)d_in[8];
    const float* g3 = (const float*)d_in[9];
    const float* b3 = (const float*)d_in[10];

    float* outXYZ  = (float*)d_out;                                   // (B,NP,3)
    float* outFeat = outXYZ + (size_t)BATCH * NPOINT * 3;             // (B,NP,128)
    float* outInds = outFeat + (size_t)BATCH * NPOINT * 128;          // (B,NP)

    char* ws = (char*)d_ws;
    int*      ballIdx = (int*)ws;                         // 512 KB
    _Float16* W1h   = (_Float16*)(ws + 524288);           // 72 KB
    _Float16* W2h   = (_Float16*)(ws + 598016);           // 32 KB
    _Float16* W3h   = (_Float16*)(ws + 630784);           // 32 KB
    _Float16* featT = (_Float16*)(ws + 663552);           // 64 MB (optional)
    const size_t needT = 663552 + (size_t)BATCH * KPTS * CFEAT * 2;
    const int useT = (ws_size >= needT) ? 1 : 0;

    fps_kernel  <<<BATCH, 256, 0, stream>>>(xyz, outXYZ, outInds);
    ballq_kernel<<<BATCH, 256, 0, stream>>>(xyz, outXYZ, ballIdx);
    wconv_kernel<<<32,    256, 0, stream>>>(W1, W2, W3, W1h, W2h, W3h);
    if (useT) {
        dim3 g(KPTS / 32, CFEAT / 32, BATCH);
        dim3 blk(32, 8, 1);
        transpose_kernel<<<g, blk, 0, stream>>>(features, featT);
    }
    mlp_kernel<<<BATCH * NPOINT / 2, 64, 0, stream>>>(
        xyz, features, featT, useT, outXYZ, ballIdx,
        W1h, g1, b1, W2h, g2, b2, W3h, g3, b3, outFeat);
}
